// NoisePredictor_73718818669271
// MI455X (gfx1250) — compile-verified, matrix operand order checked by reading, not run on hardware
//
#include <hip/hip_runtime.h>
#include <hip/hip_bf16.h>

// CDNA5 / gfx1250 PointNet++ diffusion noise predictor.
// All dense layers run through v_wmma_f32_16x16x32_f16 (wave32, f32 acc).
// GEMM operands are zero-padded in workspace (K to mult-of-32, N to mult-of-32,
// M always mult-of-16) so the WMMA inner loop has NO bounds checks:
//   A: 4x global_load_b128 (f32) + cvt, B: 2x global_load_b128 (f16), 1 wmma.

typedef __attribute__((ext_vector_type(16))) _Float16 v16h;
typedef __attribute__((ext_vector_type(8)))  _Float16 v8h;
typedef __attribute__((ext_vector_type(8)))  float    v8f;

static inline int cd(long a, long b) { return (int)((a + b - 1) / b); }
static inline int al32(int x) { return (x + 31) & ~31; }

// ---------------------------------------------------------------------------
// GEMM: Y[M,Np] = X[M,Kp] @ Wh[Np,Kp]^T + bias.  Kp%32==0, Np%16==0, M%16==0.
// One 16x16 tile per wave32, 4 waves/block.  Wh is f16, X/Y/bias f32.
// A-frag (16-bit 16x32): lane&15=row, hsel=lane>>4; per-lane halves cover
// K = k0 + hsel*8 + {0..7} and k0 + 16 + hsel*8 + {0..7}  (contiguous chunks).
// B-frag symmetric with lane&15 = output column (row of Wh).
// D: VGPR r -> M = r + 8*hsel, N = lane&15.
// ---------------------------------------------------------------------------
__global__ void k_gemm(const float* __restrict__ X, const _Float16* __restrict__ Wh,
                       const float* __restrict__ bias, float* __restrict__ Y,
                       int M, int Np, int Kp) {
  int wave  = threadIdx.x >> 5;
  int lane  = threadIdx.x & 31;
  int mtile = blockIdx.y * 4 + wave;
  int m0 = mtile * 16;
  if (m0 >= M) return;                       // uniform per wave
  int n0   = blockIdx.x * 16;
  int lr   = lane & 15;
  int hsel = lane >> 4;
  const float*    xr = X  + (long)(m0 + lr) * Kp + hsel * 8;
  const _Float16* wr = Wh + (long)(n0 + lr) * Kp + hsel * 8;
  v8f acc = {};
  for (int k0 = 0; k0 < Kp; k0 += 32) {
    float4 a0 = *(const float4*)(xr + k0);
    float4 a1 = *(const float4*)(xr + k0 + 4);
    float4 a2 = *(const float4*)(xr + k0 + 16);
    float4 a3 = *(const float4*)(xr + k0 + 20);
    v8h    b0 = *(const v8h*)(wr + k0);
    v8h    b1 = *(const v8h*)(wr + k0 + 16);
    v16h af, bf;
    af[0]  = (_Float16)a0.x; af[1]  = (_Float16)a0.y;
    af[2]  = (_Float16)a0.z; af[3]  = (_Float16)a0.w;
    af[4]  = (_Float16)a1.x; af[5]  = (_Float16)a1.y;
    af[6]  = (_Float16)a1.z; af[7]  = (_Float16)a1.w;
    af[8]  = (_Float16)a2.x; af[9]  = (_Float16)a2.y;
    af[10] = (_Float16)a2.z; af[11] = (_Float16)a2.w;
    af[12] = (_Float16)a3.x; af[13] = (_Float16)a3.y;
    af[14] = (_Float16)a3.z; af[15] = (_Float16)a3.w;
#pragma unroll
    for (int i = 0; i < 8; ++i) { bf[i] = b0[i]; bf[i + 8] = b1[i]; }
    acc = __builtin_amdgcn_wmma_f32_16x16x32_f16(false, af, false, bf,
                                                 (short)0, acc, false, false);
  }
  int n = n0 + lr;
  float bv = bias[n];
#pragma unroll
  for (int r = 0; r < 8; ++r)
    Y[(long)(m0 + r + hsel * 8) * Np + n] = acc[r] + bv;
}

// Pad+convert weights (co,ci) f32 -> (Np,Kp) f16, zero fill.
__global__ void k_padw(_Float16* __restrict__ dst, const float* __restrict__ src,
                       int co, int ci, int Np, int Kp, long total) {
  long i = blockIdx.x * (long)blockDim.x + threadIdx.x;
  if (i >= total) return;
  int k = (int)(i % Kp);
  int n = (int)(i / Kp);
  float v = (n < co && k < ci) ? src[(long)n * ci + k] : 0.f;
  dst[i] = (_Float16)v;
}

__global__ void k_padb(float* __restrict__ dst, const float* __restrict__ src,
                       int co, int Np) {
  int i = blockIdx.x * blockDim.x + threadIdx.x;
  if (i >= Np) return;
  dst[i] = (i < co) ? src[i] : 0.f;
}

// ---------------------------------------------------------------------------
// BatchNorm: folded scale/shift (two pass).  Y rows stride Np, C real channels.
// ---------------------------------------------------------------------------
__global__ void k_bn_stats(const float* __restrict__ Y, const float* __restrict__ g,
                           const float* __restrict__ be, float* __restrict__ ss,
                           long M, int C, int Np) {
  int c = blockIdx.x;            // c < C
  __shared__ float s1[256];
  __shared__ float s2[256];
  float sum = 0.f, sq = 0.f;
  for (long i = threadIdx.x; i < M; i += blockDim.x) {
    float v = Y[i * Np + c];
    sum += v; sq += v * v;
  }
  s1[threadIdx.x] = sum; s2[threadIdx.x] = sq;
  __syncthreads();
  for (int s = blockDim.x >> 1; s > 0; s >>= 1) {
    if ((int)threadIdx.x < s) {
      s1[threadIdx.x] += s1[threadIdx.x + s];
      s2[threadIdx.x] += s2[threadIdx.x + s];
    }
    __syncthreads();
  }
  if (threadIdx.x == 0) {
    float m  = s1[0] / (float)M;
    float v  = s2[0] / (float)M - m * m;
    float sc = g[c] * rsqrtf(v + 1e-5f);
    ss[2 * c]     = sc;
    ss[2 * c + 1] = be[c] - m * sc;
  }
}

// mode: 0 none, 1 relu, 2 leaky(0.2).  Pad channels (c>=C) stay zero.
__global__ void k_bn_apply(float* __restrict__ Y, const float* __restrict__ ss,
                           long total, int C, int Np, int mode) {
  long i = blockIdx.x * (long)blockDim.x + threadIdx.x;
  if (i >= total) return;
  int c = (int)(i % Np);
  if (c >= C) return;
  float v = Y[i] * ss[2 * c] + ss[2 * c + 1];
  if (mode == 1)      v = fmaxf(v, 0.f);
  else if (mode == 2) v = v > 0.f ? v : 0.2f * v;
  Y[i] = v;
}

// Fused BN + ReLU + max over group dim Kg. Y is (BS,Kg,Np); writes
// feat[bs, coff+c] with row stride Ctot.
__global__ void k_bn_apply_pool(const float* __restrict__ Y, const float* __restrict__ ss,
                                float* __restrict__ feat, int BS, int Kg, int C,
                                int Np, int Ctot, int coff) {
  int idx = blockIdx.x * blockDim.x + threadIdx.x;
  if (idx >= BS * C) return;
  int bs = idx / C, c = idx % C;
  float sc = ss[2 * c], sh = ss[2 * c + 1];
  const float* p = Y + ((long)bs * Kg) * Np + c;
  float m = -1e30f;
  for (int k = 0; k < Kg; ++k) {
    float v = fmaxf(p[(long)k * Np] * sc + sh, 0.f);
    m = fmaxf(m, v);
  }
  feat[(long)bs * Ctot + coff + c] = m;
}

// ---------------------------------------------------------------------------
// Elementwise / layout kernels
// ---------------------------------------------------------------------------
__global__ void k_transpose_in(const float* __restrict__ xt, float* __restrict__ x,
                               int B, int N) {
  int i = blockIdx.x * blockDim.x + threadIdx.x;
  if (i >= B * N * 3) return;
  int c = i % 3; int n = (i / 3) % N; int b = i / (3 * N);
  x[i] = xt[((long)b * 3 + c) * N + n];
}

// feat0[b,n,c(32)] = c<3 ? x[b,n,c] + inj[b*32+c] : 0
__global__ void k_feat0_init(float* __restrict__ feat0, const float* __restrict__ x,
                             const float* __restrict__ inj, int N, long total) {
  long i = blockIdx.x * (long)blockDim.x + threadIdx.x;
  if (i >= total) return;
  int c = (int)(i % 32);
  long bn = i / 32;
  int b = (int)(bn / N);
  feat0[i] = (c < 3) ? x[bn * 3 + c] + inj[b * 32 + c] : 0.f;
}

// et rows stride 32, real 3 channels -> out (B,3,N)
__global__ void k_transpose_out(const float* __restrict__ et, float* __restrict__ out,
                                int B, int N) {
  int i = blockIdx.x * blockDim.x + threadIdx.x;
  if (i >= B * 3 * N) return;
  int n = i % N; int c = (i / N) % 3; int b = i / (3 * N);
  out[i] = et[((long)b * N + n) * 32 + c];
}

// ctx (16,544): 0..511 z, 512..514 time_emb, rest 0
__global__ void k_ctx(const float* __restrict__ z, const float* __restrict__ te,
                      float* __restrict__ ctx, int B) {
  int i = blockIdx.x * blockDim.x + threadIdx.x;
  if (i >= B * 544) return;
  int b = i / 544, c = i % 544;
  ctx[i] = (c < 512) ? z[b * 512 + c] : (c < 515 ? te[b * 3 + (c - 512)] : 0.f);
}

// dst[b,n,c] = src[b,n,c] + inj[b,c]; strides Cp (pads are 0+0). May alias.
__global__ void k_bcast_sum(float* __restrict__ dst, const float* __restrict__ src,
                            const float* __restrict__ inj, int Nn, int Cp, long total) {
  long i = blockIdx.x * (long)blockDim.x + threadIdx.x;
  if (i >= total) return;
  int c = (int)(i % Cp);
  int b = (int)(i / ((long)Nn * Cp));
  dst[i] = src[i] + inj[b * Cp + c];
}

__global__ void k_add(float* __restrict__ a, const float* __restrict__ b, long total) {
  long i = blockIdx.x * (long)blockDim.x + threadIdx.x;
  if (i < total) a[i] += b[i];
}

// ---------------------------------------------------------------------------
// Geometry kernels (unchanged semantics)
// ---------------------------------------------------------------------------
__global__ void k_fps(const float* __restrict__ xyz, int* __restrict__ idx, int N, int S) {
  int b = blockIdx.x;
  const float* p = xyz + (long)b * N * 3;
  __shared__ float sv[256];
  __shared__ int   si[256];
  __shared__ int   sfar;
  float dist[8];
  int cnt = 0;
  for (int n = threadIdx.x; n < N; n += blockDim.x) dist[cnt++] = 1e10f;
  if (threadIdx.x == 0) sfar = 0;
  __syncthreads();
  for (int it = 0; it < S; ++it) {
    int far = sfar;
    if (threadIdx.x == 0) idx[b * S + it] = far;
    float cx = p[far * 3], cy = p[far * 3 + 1], cz = p[far * 3 + 2];
    float best = -1.f; int bi = 0x7fffffff; cnt = 0;
    for (int n = threadIdx.x; n < N; n += blockDim.x) {
      float dx = p[n * 3] - cx, dy = p[n * 3 + 1] - cy, dz = p[n * 3 + 2] - cz;
      float d = dx * dx + dy * dy + dz * dz;
      float dm = fminf(dist[cnt], d);
      dist[cnt++] = dm;
      if (dm > best) { best = dm; bi = n; }
    }
    sv[threadIdx.x] = best; si[threadIdx.x] = bi;
    __syncthreads();
    for (int s = blockDim.x >> 1; s > 0; s >>= 1) {
      if ((int)threadIdx.x < s) {
        float vo = sv[threadIdx.x + s]; int io = si[threadIdx.x + s];
        if (vo > sv[threadIdx.x] || (vo == sv[threadIdx.x] && io < si[threadIdx.x])) {
          sv[threadIdx.x] = vo; si[threadIdx.x] = io;
        }
      }
      __syncthreads();
    }
    if (threadIdx.x == 0) sfar = si[0];
    __syncthreads();
  }
}

__global__ void k_gather_xyz(const float* __restrict__ xyz, const int* __restrict__ fi,
                             float* __restrict__ nxyz, int N, int S, int BS) {
  int i = blockIdx.x * blockDim.x + threadIdx.x;
  if (i >= BS) return;
  int b = i / S;
  int g = fi[i];
  const float* p = xyz + ((long)b * N + g) * 3;
  nxyz[(long)i * 3]     = p[0];
  nxyz[(long)i * 3 + 1] = p[1];
  nxyz[(long)i * 3 + 2] = p[2];
}

__global__ void k_ball(const float* __restrict__ xyz, const float* __restrict__ nxyz,
                       int* __restrict__ gi, int N, int S, int Kg, float r2, int BS) {
  int i = blockIdx.x * blockDim.x + threadIdx.x;
  if (i >= BS) return;
  int b = i / S;
  const float* p = xyz + (long)b * N * 3;
  float cx = nxyz[(long)i * 3], cy = nxyz[(long)i * 3 + 1], cz = nxyz[(long)i * 3 + 2];
  int* out = gi + (long)i * Kg;
  int cnt = 0, first = -1;
  for (int n = 0; n < N && cnt < Kg; ++n) {
    float dx = p[n * 3] - cx, dy = p[n * 3 + 1] - cy, dz = p[n * 3 + 2] - cz;
    if (dx * dx + dy * dy + dz * dz <= r2) {
      if (first < 0) first = n;
      out[cnt++] = n;
    }
  }
  if (first < 0) first = 0;
  for (; cnt < Kg; ++cnt) out[cnt] = first;
}

// grouped[b,s,k, 0:C]=feat[gi] (stride Cfp), [C:C+3]=xyz[gi]-new_xyz, pad 0.
__global__ void k_group(const float* __restrict__ xyz, const float* __restrict__ feat,
                        const float* __restrict__ nxyz, const int* __restrict__ gi,
                        float* __restrict__ out, int N, int S, int Kg,
                        int C, int Cfp, int Cinp, long total) {
  long i = blockIdx.x * (long)blockDim.x + threadIdx.x;
  if (i >= total) return;
  int c = (int)(i % Cinp);
  long r = i / Cinp;           // b*S*Kg + s*Kg + k
  long bs = r / Kg;            // b*S + s
  int b = (int)(bs / S);
  int g = gi[r];
  float v;
  if (c < C)          v = feat[((long)b * N + g) * Cfp + c];
  else if (c < C + 3) v = xyz[((long)b * N + g) * 3 + (c - C)] - nxyz[bs * 3 + (c - C)];
  else                v = 0.f;
  out[i] = v;
}

__global__ void k_knn3(const float* __restrict__ xyz1, const float* __restrict__ xyz2,
                       int* __restrict__ ni, float* __restrict__ wgt,
                       int N1, int N2, int BN1) {
  int i = blockIdx.x * blockDim.x + threadIdx.x;
  if (i >= BN1) return;
  int b = i / N1;
  const float* p2 = xyz2 + (long)b * N2 * 3;
  float x = xyz1[(long)i * 3], y = xyz1[(long)i * 3 + 1], zc = xyz1[(long)i * 3 + 2];
  float d0 = 1e30f, d1 = 1e30f, d2 = 1e30f;
  int   i0 = 0, i1 = 0, i2 = 0;
  for (int n = 0; n < N2; ++n) {
    float dx = p2[n * 3] - x, dy = p2[n * 3 + 1] - y, dz = p2[n * 3 + 2] - zc;
    float d = dx * dx + dy * dy + dz * dz;
    if (d < d0)      { d2 = d1; i2 = i1; d1 = d0; i1 = i0; d0 = d; i0 = n; }
    else if (d < d1) { d2 = d1; i2 = i1; d1 = d;  i1 = n; }
    else if (d < d2) { d2 = d;  i2 = n; }
  }
  float w0 = 1.f / (d0 + 1e-8f), w1 = 1.f / (d1 + 1e-8f), w2 = 1.f / (d2 + 1e-8f);
  float s = w0 + w1 + w2;
  ni[(long)i * 3] = i0;  ni[(long)i * 3 + 1] = i1;  ni[(long)i * 3 + 2] = i2;
  wgt[(long)i * 3] = w0 / s; wgt[(long)i * 3 + 1] = w1 / s; wgt[(long)i * 3 + 2] = w2 / s;
}

// h[b,n, 0:C1]=f1 (stride C1p), [C1:C1+C2]=interp(f2, stride C2p), pad 0.
__global__ void k_fp_build(const float* __restrict__ f1, const float* __restrict__ f2,
                           const int* __restrict__ ni, const float* __restrict__ wgt,
                           float* __restrict__ h, int N1, int N2,
                           int C1, int C1p, int C2, int C2p, int Cp, long total) {
  long i = blockIdx.x * (long)blockDim.x + threadIdx.x;
  if (i >= total) return;
  int c = (int)(i % Cp);
  long r = i / Cp;             // b*N1 + n
  int b = (int)(r / N1);
  float v;
  if (c < C1) v = f1[r * C1p + c];
  else if (c < C1 + C2) {
    int cc = c - C1;
    const float* fb = f2 + (long)b * N2 * C2p;
    v = wgt[r * 3]     * fb[(long)ni[r * 3]     * C2p + cc]
      + wgt[r * 3 + 1] * fb[(long)ni[r * 3 + 1] * C2p + cc]
      + wgt[r * 3 + 2] * fb[(long)ni[r * 3 + 2] * C2p + cc];
  } else v = 0.f;
  h[i] = v;
}

// Attention scores + softmax: one block of S(=32) threads per (b, qi).
__global__ void k_attn_scores(const float* __restrict__ q, const float* __restrict__ kb,
                              float* __restrict__ a, int S, int C) {
  int b = blockIdx.x / S, qi = blockIdx.x % S;
  int t = threadIdx.x;
  const float* qp = q  + ((long)b * S + qi) * C;
  const float* kp = kb + ((long)b * S + t) * C;
  float s = 0.f;
  for (int c = 0; c < C; ++c) s += qp[c] * kp[c];
  s *= rsqrtf((float)C);
  __shared__ float sh[64];
  sh[t] = s; __syncthreads();
  float mx = -1e30f;
  for (int j = 0; j < S; ++j) mx = fmaxf(mx, sh[j]);
  __syncthreads();
  float e = expf(s - mx);
  sh[t] = e; __syncthreads();
  float sum = 0.f;
  for (int j = 0; j < S; ++j) sum += sh[j];
  a[((long)b * S + qi) * S + t] = e / sum;
}

__global__ void k_attn_av(const float* __restrict__ a, const float* __restrict__ v,
                          float* __restrict__ o, int S, int C, long total) {
  long i = blockIdx.x * (long)blockDim.x + threadIdx.x;
  if (i >= total) return;
  int c = (int)(i % C);
  long r = i / C;              // b*S + qi
  int b = (int)(r / S);
  const float* ap = a + r * S;
  const float* vp = v + (long)b * S * C;
  float s = 0.f;
  for (int k = 0; k < S; ++k) s += ap[k] * vp[(long)k * C + c];
  o[i] = s;
}

// ---------------------------------------------------------------------------
// Host-side orchestration
// ---------------------------------------------------------------------------
struct GL { const _Float16* w; const float* b; const float* g; const float* be;
            int co, ci, Np, Kp; };

static void gemm(hipStream_t st, const float* X, const GL& L, float* Y, long M) {
  dim3 grid(L.Np / 16, cd(M, 64));
  k_gemm<<<grid, dim3(128), 0, st>>>(X, L.w, L.b, Y, (int)M, L.Np, L.Kp);
}

static void lin_bn_act(hipStream_t st, const float* X, const GL& L, float* Y,
                       float* ss, long M, int mode) {
  gemm(st, X, L, Y, M);
  k_bn_stats<<<dim3(L.co), dim3(256), 0, st>>>(Y, L.g, L.be, ss, M, L.co, L.Np);
  long tot = M * L.Np;
  k_bn_apply<<<dim3(cd(tot, 256)), dim3(256), 0, st>>>(Y, ss, tot, L.co, L.Np, mode);
}

static void msg_branch(hipStream_t st, const float* xyz, const float* feat,
                       int C, int Cfp, const float* nxyz, int N, int S,
                       float radius, int Kg, const GL* L, int nl,
                       float* feat_out, int Ctot, int coff,
                       int* gi, float* bufG, float* bufA, float* bufB, float* ss) {
  int BS = 16 * S;
  k_ball<<<dim3(cd(BS, 256)), dim3(256), 0, st>>>(xyz, nxyz, gi, N, S, Kg,
                                                  radius * radius, BS);
  long M = (long)BS * Kg;
  int Cinp = L[0].Kp;                       // = al32(C+3)
  long totG = M * Cinp;
  k_group<<<dim3(cd(totG, 256)), dim3(256), 0, st>>>(xyz, feat, nxyz, gi, bufG,
                                                     N, S, Kg, C, Cfp, Cinp, totG);
  const float* in = bufG;
  float* out = bufA;
  for (int l = 0; l < nl - 1; ++l) {
    lin_bn_act(st, in, L[l], out, ss, M, 1);
    in = out;
    out = (out == bufA) ? bufB : bufA;
  }
  const GL& Lf = L[nl - 1];
  gemm(st, in, Lf, out, M);
  k_bn_stats<<<dim3(Lf.co), dim3(256), 0, st>>>(out, Lf.g, Lf.be, ss, M, Lf.co, Lf.Np);
  k_bn_apply_pool<<<dim3(cd((long)BS * Lf.co, 256)), dim3(256), 0, st>>>(
      out, ss, feat_out, BS, Kg, Lf.co, Lf.Np, Ctot, coff);
}

static void fp_stage(hipStream_t st, const float* xyz_a, int N1, const float* xyz_b, int N2,
                     const float* f1, int C1, int C1p, const float* f2, int C2, int C2p,
                     const GL* L, int nl, float* out_feat,
                     int* ni, float* wgt, float* bufH, float* bufA, float* bufB, float* ss) {
  int BN1 = 16 * N1;
  k_knn3<<<dim3(cd(BN1, 256)), dim3(256), 0, st>>>(xyz_a, xyz_b, ni, wgt, N1, N2, BN1);
  int Cp = L[0].Kp;                         // = al32(C1+C2)
  long tot = (long)BN1 * Cp;
  k_fp_build<<<dim3(cd(tot, 256)), dim3(256), 0, st>>>(f1, f2, ni, wgt, bufH,
                                                       N1, N2, C1, C1p, C2, C2p, Cp, tot);
  const float* in = bufH;
  long M = BN1;
  for (int l = 0; l < nl; ++l) {
    float* o = (l == nl - 1) ? out_feat : ((l & 1) ? bufB : bufA);
    lin_bn_act(st, in, L[l], o, ss, M, 1);
    in = o;
  }
}

extern "C" void kernel_launch(void* const* d_in, const int* in_sizes, int n_in,
                              void* d_out, int out_size, void* d_ws, size_t ws_size,
                              hipStream_t stream) {
  (void)in_sizes; (void)n_in; (void)out_size; (void)ws_size;
  const int B = 16, N = 2048;

  // ---- workspace bump allocator -------------------------------------------
  char* base = (char*)d_ws;
  size_t off = 0;
  auto allocf = [&](size_t n) -> float* {
    float* q = (float*)(base + off);
    off = (off + n * sizeof(float) + 255) & ~(size_t)255;
    return q;
  };
  auto alloch = [&](size_t n) -> _Float16* {
    _Float16* q = (_Float16*)(base + off);
    off = (off + n * sizeof(_Float16) + 255) & ~(size_t)255;
    return q;
  };
  auto alloci = [&](size_t n) -> int* {
    int* q = (int*)(base + off);
    off = (off + n * sizeof(int) + 255) & ~(size_t)255;
    return q;
  };

  // ---- unpack inputs (jax tree_leaves order: dict keys sorted) -------------
  // top: params(an1..6, attn1, down1..3, pred, up1..3), time_emb, xt, z
  int p = 0;
  auto nxt = [&]() { return (const float*)d_in[p++]; };
  auto mk = [&](const float* w, const float* b, int co, int ci,
                const float* g, const float* be) -> GL {
    GL L; L.co = co; L.ci = ci; L.Np = al32(co); L.Kp = al32(ci);
    _Float16* wp = alloch((size_t)L.Np * L.Kp);
    float*    bp = allocf(L.Np);
    long tw = (long)L.Np * L.Kp;
    k_padw<<<dim3(cd(tw, 256)), dim3(256), 0, stream>>>(wp, w, co, ci, L.Np, L.Kp, tw);
    k_padb<<<dim3(cd(L.Np, 256)), dim3(256), 0, stream>>>(bp, b, co, L.Np);
    L.w = wp; L.b = bp; L.g = g; L.be = be;
    return L;
  };

  struct InjP { GL l1, l2; int dout; };
  InjP an[6]; const int andim[6] = {3, 96, 256, 512, 256, 128};
  for (int i = 0; i < 6; ++i) {            // an*: be, g, l1{b,w}, l2{b,w}
    const float* be  = nxt(); const float* g   = nxt();
    const float* l1b = nxt(); const float* l1w = nxt();
    const float* l2b = nxt(); const float* l2w = nxt();
    an[i].l1 = mk(l1w, l1b, 512, 515, g, be);
    an[i].l2 = mk(l2w, l2b, andim[i], 512, nullptr, nullptr);
    an[i].dout = andim[i];
  }
  // attn1: wk, wo, wq, wv ; each {b,w}
  const float *wkb = nxt(), *wkw = nxt(), *wob = nxt(), *wow = nxt();
  const float *wqb = nxt(), *wqw = nxt(), *wvb = nxt(), *wvw = nxt();
  GL kL = mk(wkw, wkb, 512, 512, nullptr, nullptr);
  GL oL = mk(wow, wob, 512, 512, nullptr, nullptr);
  GL qL = mk(wqw, wqb, 512, 512, nullptr, nullptr);
  GL vL = mk(wvw, wvb, 512, 512, nullptr, nullptr);
  auto rdl = [&](int ci, int co) -> GL {   // layer: b, be, g, w
    const float* b  = nxt(); const float* be = nxt();
    const float* g  = nxt(); const float* w  = nxt();
    return mk(w, b, co, ci, g, be);
  };
  GL d1a[3] = { rdl(6, 16),    rdl(16, 16),   rdl(16, 32)   };
  GL d1b[3] = { rdl(6, 32),    rdl(32, 32),   rdl(32, 64)   };
  GL d2a[3] = { rdl(99, 64),   rdl(64, 64),   rdl(64, 128)  };
  GL d2b[3] = { rdl(99, 64),   rdl(64, 96),   rdl(96, 128)  };
  GL d3a[3] = { rdl(259, 128), rdl(128, 196), rdl(196, 256) };
  GL d3b[3] = { rdl(259, 128), rdl(128, 196), rdl(196, 256) };
  // pred: be, c1{b,w}, c2{b,w}, g
  const float* pred_be = nxt();
  const float* c1b = nxt(); const float* c1w = nxt();
  const float* c2b = nxt(); const float* c2w = nxt();
  const float* pred_g = nxt();
  GL c1L = mk(c1w, c1b, 512, 256, pred_g, pred_be);
  GL c2L = mk(c2w, c2b, 3, 512, nullptr, nullptr);
  GL up1[2] = { rdl(768, 256), rdl(256, 256) };
  GL up2[3] = { rdl(352, 256), rdl(256, 128), rdl(128, 128) };
  GL up3[3] = { rdl(131, 128), rdl(128, 128), rdl(128, 256) };
  const float* time_emb = nxt();
  const float* xt       = nxt();
  const float* z        = nxt();

  // ---- activation workspace ------------------------------------------------
  float* ctx    = allocf((size_t)16 * 544);
  float* x      = allocf((size_t)B * N * 3);
  float* feat0  = allocf((size_t)B * N * 32);     // 3 real, stride 32
  float* injh   = allocf(16 * 512);
  float* injo   = allocf(16 * 512);               // stride = al32(dout)
  float* ss     = allocf(1024);
  int*   fpsidx = alloci(16 * 1024);
  float* xyz1   = allocf(16 * 1024 * 3);
  float* xyz2   = allocf(16 * 256 * 3);
  float* xyz3   = allocf(16 * 32 * 3);
  float* feat1  = allocf((size_t)16 * 1024 * 96);
  float* feat2  = allocf((size_t)16 * 256 * 256);
  float* feat3  = allocf((size_t)16 * 32 * 512);
  int*   gi     = alloci((size_t)16 * 1024 * 32);
  int*   nni    = alloci((size_t)B * N * 3);
  float* wknn   = allocf((size_t)B * N * 3);
  float* qb     = allocf(16 * 32 * 512);
  float* kb     = allocf(16 * 32 * 512);
  float* vb     = allocf(16 * 32 * 512);
  float* avb    = allocf(16 * 32 * 512);
  float* scb    = allocf(16 * 32 * 32);
  float* f3n    = allocf(16 * 32 * 512);
  float* feat2n = allocf((size_t)16 * 256 * 256);
  float* feat1n = allocf((size_t)16 * 1024 * 128);
  float* feat0n = allocf((size_t)16 * 2048 * 256);
  float* bufG   = allocf(33600000);   // >= down2b grouped 16*256*64*128
  float* bufA   = allocf(33600000);   // >= 16*1024*32*64, 16*256*64*128
  float* bufB   = allocf(33600000);

  auto inj = [&](const InjP& I, float* o) {
    gemm(stream, ctx, I.l1, injh, 16);
    k_bn_stats<<<dim3(512), dim3(256), 0, stream>>>(injh, I.l1.g, I.l1.be, ss, 16, 512, 512);
    k_bn_apply<<<dim3(cd(16 * 512, 256)), dim3(256), 0, stream>>>(
        injh, ss, 16 * 512, 512, 512, 1);
    gemm(stream, injh, I.l2, o, 16);
  };

  // ---- forward -------------------------------------------------------------
  k_ctx<<<dim3(cd(16 * 544, 256)), dim3(256), 0, stream>>>(z, time_emb, ctx, B);
  k_transpose_in<<<dim3(cd((long)B * N * 3, 256)), dim3(256), 0, stream>>>(xt, x, B, N);

  inj(an[0], injo);
  k_feat0_init<<<dim3(cd((long)B * N * 32, 256)), dim3(256), 0, stream>>>(
      feat0, x, injo, N, (long)B * N * 32);

  // down1: 2048 -> 1024  (feat stride 32, real C=3; out feat1 stride 96)
  k_fps<<<dim3(16), dim3(256), 0, stream>>>(x, fpsidx, N, 1024);
  k_gather_xyz<<<dim3(cd(16 * 1024, 256)), dim3(256), 0, stream>>>(
      x, fpsidx, xyz1, N, 1024, 16 * 1024);
  msg_branch(stream, x, feat0, 3, 32, xyz1, N, 1024, 0.1f, 16, d1a, 3, feat1, 96, 0,
             gi, bufG, bufA, bufB, ss);
  msg_branch(stream, x, feat0, 3, 32, xyz1, N, 1024, 0.3f, 32, d1b, 3, feat1, 96, 32,
             gi, bufG, bufA, bufB, ss);
  inj(an[1], injo);
  k_bcast_sum<<<dim3(cd((long)16 * 1024 * 96, 256)), dim3(256), 0, stream>>>(
      feat1, feat1, injo, 1024, 96, (long)16 * 1024 * 96);

  // down2: 1024 -> 256
  k_fps<<<dim3(16), dim3(256), 0, stream>>>(xyz1, fpsidx, 1024, 256);
  k_gather_xyz<<<dim3(cd(16 * 256, 256)), dim3(256), 0, stream>>>(
      xyz1, fpsidx, xyz2, 1024, 256, 16 * 256);
  msg_branch(stream, xyz1, feat1, 96, 96, xyz2, 1024, 256, 0.3f, 32, d2a, 3, feat2, 256, 0,
             gi, bufG, bufA, bufB, ss);
  msg_branch(stream, xyz1, feat1, 96, 96, xyz2, 1024, 256, 0.5f, 64, d2b, 3, feat2, 256, 128,
             gi, bufG, bufA, bufB, ss);
  inj(an[2], injo);
  k_bcast_sum<<<dim3(cd((long)16 * 256 * 256, 256)), dim3(256), 0, stream>>>(
      feat2, feat2, injo, 256, 256, (long)16 * 256 * 256);

  // down3: 256 -> 32
  k_fps<<<dim3(16), dim3(256), 0, stream>>>(xyz2, fpsidx, 256, 32);
  k_gather_xyz<<<dim3(cd(16 * 32, 256)), dim3(256), 0, stream>>>(
      xyz2, fpsidx, xyz3, 256, 32, 16 * 32);
  msg_branch(stream, xyz2, feat2, 256, 256, xyz3, 256, 32, 0.5f, 64, d3a, 3, feat3, 512, 0,
             gi, bufG, bufA, bufB, ss);
  msg_branch(stream, xyz2, feat2, 256, 256, xyz3, 256, 32, 0.7f, 128, d3b, 3, feat3, 512, 256,
             gi, bufG, bufA, bufB, ss);

  // attention on feat3 (16,32,512)
  gemm(stream, feat3, qL, qb, 512);
  gemm(stream, feat3, kL, kb, 512);
  gemm(stream, feat3, vL, vb, 512);
  k_attn_scores<<<dim3(16 * 32), dim3(32), 0, stream>>>(qb, kb, scb, 32, 512);
  k_attn_av<<<dim3(cd((long)16 * 32 * 512, 256)), dim3(256), 0, stream>>>(
      scb, vb, avb, 32, 512, (long)16 * 32 * 512);
  gemm(stream, avb, oL, f3n, 512);
  k_add<<<dim3(cd((long)16 * 32 * 512, 256)), dim3(256), 0, stream>>>(
      f3n, feat3, (long)16 * 32 * 512);
  inj(an[3], injo);
  k_bcast_sum<<<dim3(cd((long)16 * 32 * 512, 256)), dim3(256), 0, stream>>>(
      f3n, f3n, injo, 32, 512, (long)16 * 32 * 512);

  // up1: interp to 256 pts, [768->256->256]
  fp_stage(stream, xyz2, 256, xyz3, 32, feat2, 256, 256, f3n, 512, 512, up1, 2, feat2n,
           nni, wknn, bufG, bufA, bufB, ss);
  inj(an[4], injo);
  k_bcast_sum<<<dim3(cd((long)16 * 256 * 256, 256)), dim3(256), 0, stream>>>(
      feat2n, feat2n, injo, 256, 256, (long)16 * 256 * 256);

  // up2: -> 1024 pts, [352->256->128->128]
  fp_stage(stream, xyz1, 1024, xyz2, 256, feat1, 96, 96, feat2n, 256, 256, up2, 3, feat1n,
           nni, wknn, bufG, bufA, bufB, ss);
  inj(an[5], injo);
  k_bcast_sum<<<dim3(cd((long)16 * 1024 * 128, 256)), dim3(256), 0, stream>>>(
      feat1n, feat1n, injo, 1024, 128, (long)16 * 1024 * 128);

  // up3: -> 2048 pts, [131->128->128->256]  (f1 = feat0: real 3, stride 32)
  fp_stage(stream, x, 2048, xyz1, 1024, feat0, 3, 32, feat1n, 128, 128, up3, 3, feat0n,
           nni, wknn, bufG, bufA, bufB, ss);

  // pred head: 256 -> 512 (BN + leaky 0.2) -> 3 (padded to 32)
  long Mp = (long)B * N;
  gemm(stream, feat0n, c1L, bufA, Mp);
  k_bn_stats<<<dim3(512), dim3(256), 0, stream>>>(bufA, pred_g, pred_be, ss, Mp, 512, 512);
  k_bn_apply<<<dim3(cd(Mp * 512, 256)), dim3(256), 0, stream>>>(
      bufA, ss, Mp * 512, 512, 512, 2);
  gemm(stream, bufA, c2L, bufB, Mp);
  k_transpose_out<<<dim3(cd((long)B * 3 * N, 256)), dim3(256), 0, stream>>>(
      bufB, (float*)d_out, B, N);
}
